// MultiHeadSelfAttention_65481071399163
// MI455X (gfx1250) — matrix operand order repaired and checked by reading, not compiled, not run on hardware
//
#include <hip/hip_runtime.h>
#include <stdint.h>

#define BATCH 2
#define SEQ   2048
#define DIM   1024
#define HEADS 16
#define HDIM  64
#define MROWS (BATCH * SEQ)   // 4096

typedef __attribute__((ext_vector_type(16))) __bf16 v16bf;
typedef __attribute__((ext_vector_type(8)))  float  v8f;
typedef int v4i __attribute__((vector_size(16)));

struct __align__(16) U4 { unsigned int x, y, z, w; };

union Frag16 {
  U4 q2[2];
  unsigned short h[16];
  v16bf v;
};

__device__ __forceinline__ unsigned short f2bf(float f) {
  unsigned int u = __builtin_bit_cast(unsigned int, f);
  unsigned int r = (u + 0x7FFFu + ((u >> 16) & 1u)) >> 16;
  return (unsigned short)r;
}

__device__ __forceinline__ v8f wmma_bf16(const Frag16& a, const Frag16& b, v8f c) {
  return __builtin_amdgcn_wmma_f32_16x16x32_bf16(false, a.v, false, b.v,
                                                 (short)0, c, false, false);
}

// ---- optional CDNA5 async global->LDS path (probe-guarded) ----------------
#if defined(__has_builtin)
#  if __has_builtin(__builtin_amdgcn_global_load_async_to_lds_b128) && \
      __has_builtin(__builtin_amdgcn_s_wait_asynccnt)
#    define USE_ASYNC_LDS 1
#  endif
#endif
#ifndef USE_ASYNC_LDS
#  define USE_ASYNC_LDS 0
#endif

#if USE_ASYNC_LDS
__device__ __forceinline__ void async_copy_b128(const void* gsrc, void* ldst) {
  __builtin_amdgcn_global_load_async_to_lds_b128(
      (__attribute__((address_space(1))) v4i*)gsrc,
      (__attribute__((address_space(3))) v4i*)ldst, 0, 0);
}
#endif

// ---------------------------------------------------------------- convert
__global__ __launch_bounds__(256)
void cvt_f32_bf16_x8(const float* __restrict__ in,
                     unsigned short* __restrict__ out, int n8) {
  const int i = blockIdx.x * blockDim.x + threadIdx.x;
  if (i >= n8) return;
  const float* p = in + (size_t)i * 8;
  union { U4 q; unsigned short h[8]; } r;
#pragma unroll
  for (int j = 0; j < 8; ++j) r.h[j] = f2bf(p[j]);
  *(U4*)(out + (size_t)i * 8) = r.q;
}

// ---------------------------------------------------------------- GEMM
// C[M,N] = A[M,K](bf16) @ B[K,N](bf16) + bias
// MODE 0: out bf16 scattered to (B,H,S,Hd) layout, scaled
// MODE 1: out fp32 row-major
#define BM 128
#define BN 128
#define BKK 64
#define LDT 72   // LDS row stride in halves: 144 B, 16B-aligned

template<int MODE>
__global__ __launch_bounds__(256)
void gemm_bf16_wmma(const unsigned short* __restrict__ A,
                    const unsigned short* __restrict__ Bw,
                    const float* __restrict__ bias,
                    float* __restrict__ outF,
                    unsigned short* __restrict__ outQ,
                    float scale) {
  constexpr int K = DIM;
  constexpr int N = DIM;
  __shared__ __align__(16) unsigned short As[BM][LDT];
  __shared__ __align__(16) unsigned short Bs[BN][LDT];   // transposed: [n][k]

  const int tid   = threadIdx.x;
  const int wid   = tid >> 5;
  const int lane  = tid & 31;
  const int lan16 = lane & 15;
  const int half  = lane >> 4;
  const int m0 = blockIdx.y * BM;
  const int n0 = blockIdx.x * BN;
  const int wm = (wid >> 2) * 64;
  const int wn = (wid & 3) * 32;

  const int arow = tid >> 1;
  const int acol = (tid & 1) * 32;
  const int bkr  = tid >> 2;
  const int bnc  = (tid & 3) * 32;

  v8f acc[4][2];
#pragma unroll
  for (int i = 0; i < 4; ++i)
#pragma unroll
    for (int j = 0; j < 2; ++j)
      acc[i][j] = (v8f){0.f,0.f,0.f,0.f,0.f,0.f,0.f,0.f};

  const int g   = half * 8;    // A-frag: K groups at {g, g+16} within 32-block
  const int kb0 = half * 8;    // B-frag: K groups at {kb0, kb0+16} within 32-block

  const unsigned short* aglob = A + (size_t)(m0 + arow) * K + acol;
  const unsigned short* bglob = Bw + (size_t)bkr * N + (n0 + bnc);

  for (int k0 = 0; k0 < K; k0 += BKK) {
    __syncthreads();
    // ---- A tile (BM x 64) row-major into LDS
#if USE_ASYNC_LDS
    {
      const unsigned short* ag = aglob + k0;
      unsigned short* ld = &As[arow][acol];
#pragma unroll
      for (int j = 0; j < 4; ++j)
        async_copy_b128(ag + j * 8, ld + j * 8);
    }
#else
    {
      const U4* ag = (const U4*)(aglob + k0);
      U4 t0 = ag[0], t1 = ag[1], t2 = ag[2], t3 = ag[3];
      U4* ld = (U4*)&As[arow][acol];
      ld[0] = t0; ld[1] = t1; ld[2] = t2; ld[3] = t3;
    }
#endif
    // ---- B tile (64 x BN), store transposed Bs[n][k]
    {
      const U4* bg = (const U4*)(bglob + (size_t)k0 * N);
      union { U4 q[4]; unsigned short h[32]; } bu;
      bu.q[0] = bg[0]; bu.q[1] = bg[1]; bu.q[2] = bg[2]; bu.q[3] = bg[3];
#pragma unroll
      for (int j = 0; j < 32; ++j)
        Bs[bnc + j][bkr] = bu.h[j];
    }
#if USE_ASYNC_LDS
    __builtin_amdgcn_s_wait_asynccnt(0);
#endif
    __syncthreads();

    // ---- 2 x (K=32) WMMA steps = 16 WMMAs per barrier pair
#pragma unroll
    for (int kb = 0; kb < 2; ++kb) {
      Frag16 af[4], bf[2];
#pragma unroll
      for (int i = 0; i < 4; ++i) {
        const unsigned short* p = &As[wm + i * 16 + lan16][kb * 32 + g];
        af[i].q2[0] = *(const U4*)p;
        af[i].q2[1] = *(const U4*)(p + 16);
      }
#pragma unroll
      for (int j = 0; j < 2; ++j) {
        const unsigned short* p = &Bs[wn + j * 16 + lan16][kb * 32 + kb0];
        bf[j].q2[0] = *(const U4*)p;
        bf[j].q2[1] = *(const U4*)(p + 16);
      }
#pragma unroll
      for (int i = 0; i < 4; ++i)
#pragma unroll
        for (int j = 0; j < 2; ++j)
          acc[i][j] = wmma_bf16(af[i], bf[j], acc[i][j]);
    }
  }

  // ---- epilogue (MODE folded at compile time, base address hoisted)
#pragma unroll
  for (int j = 0; j < 2; ++j) {
    const int col = n0 + wn + j * 16 + lan16;
    const float bv = bias[col];
#pragma unroll
    for (int i = 0; i < 4; ++i) {
      const int row0 = m0 + wm + i * 16 + half * 8;
      if (MODE == 0) {
        const int bb = row0 >> 11;
        const int ss = row0 & (SEQ - 1);
        const int hh = col >> 6;
        const int hd = col & (HDIM - 1);
        unsigned short* op =
            outQ + (((size_t)bb * HEADS + hh) * SEQ + ss) * HDIM + hd;
#pragma unroll
        for (int r = 0; r < 8; ++r)
          op[(size_t)r * HDIM] = f2bf((acc[i][j][r] + bv) * scale);
      } else {
        float* op = outF + (size_t)row0 * N + col;
#pragma unroll
        for (int r = 0; r < 8; ++r)
          op[(size_t)r * N] = acc[i][j][r] + bv;
      }
    }
  }
}

// ---------------------------------------------------------------- attention
// grid (SEQ/64, B*H), block 128 (4 waves, 16 q-rows each). Flash-style.
__global__ __launch_bounds__(128)
void attn_flash_wmma(const unsigned short* __restrict__ Qb,
                     const unsigned short* __restrict__ Kb,
                     const unsigned short* __restrict__ Vb,
                     const float* __restrict__ x,
                     unsigned short* __restrict__ cb) {
  __shared__ __align__(16) unsigned short Vt[HDIM][72];   // [hd][key]
  __shared__ __align__(16) unsigned short Pw[4][16][72];  // per-wave P

  const int tid   = threadIdx.x;
  const int wid   = tid >> 5;
  const int lane  = tid & 31;
  const int lan16 = lane & 15;
  const int half  = lane >> 4;
  const int g     = half * 8;

  const int bh = blockIdx.y;
  const int b  = bh >> 4;
  const int h  = bh & (HEADS - 1);
  const int q0 = blockIdx.x * 64;
  const size_t base = (size_t)bh * SEQ * HDIM;

  // Q fragments (rows = q0 + wid*16 .. +15); Q already carries 0.125 scale
  Frag16 qf[2];
  {
    const unsigned short* qp = Qb + base + (size_t)(q0 + wid * 16 + lan16) * HDIM;
#pragma unroll
    for (int kk = 0; kk < 2; ++kk) {
      const unsigned short* p = qp + kk * 32;
      qf[kk].q2[0] = *(const U4*)(p + g);
      qf[kk].q2[1] = *(const U4*)(p + g + 16);
    }
  }

  float mrow[8], lrow[8];
  v8f o[4];
#pragma unroll
  for (int r = 0; r < 8; ++r) { mrow[r] = -1e30f; lrow[r] = 0.f; }
#pragma unroll
  for (int d = 0; d < 4; ++d) o[d] = (v8f){0.f,0.f,0.f,0.f,0.f,0.f,0.f,0.f};

  const int vkey = tid >> 1;
  const int vhd0 = (tid & 1) * 32;

  for (int t0 = 0; t0 < SEQ; t0 += 64) {
    __syncthreads();   // previous PV reads of Vt complete
    {   // stage V tile transposed: Vt[hd][key]
      const U4* vp = (const U4*)(Vb + base + (size_t)(t0 + vkey) * HDIM + vhd0);
      union { U4 q[4]; unsigned short h[32]; } vu;
      vu.q[0] = vp[0]; vu.q[1] = vp[1]; vu.q[2] = vp[2]; vu.q[3] = vp[3];
#pragma unroll
      for (int j = 0; j < 32; ++j)
        Vt[vhd0 + j][vkey] = vu.h[j];
    }

    // scores 16x64: B-frag = kd groups {8h, 8h+16} of one K row (global)
    v8f sc[4];
#pragma unroll
    for (int nf = 0; nf < 4; ++nf) {
      sc[nf] = (v8f){0.f,0.f,0.f,0.f,0.f,0.f,0.f,0.f};
#pragma unroll
      for (int kb = 0; kb < 2; ++kb) {
        Frag16 kf;
        const unsigned short* kp = Kb + base +
            (size_t)(t0 + nf * 16 + lan16) * HDIM + kb * 32 + half * 8;
        kf.q2[0] = *(const U4*)kp;
        kf.q2[1] = *(const U4*)(kp + 16);
        sc[nf] = wmma_bf16(qf[kb], kf, sc[nf]);
      }
    }

    // online softmax: row reductions across the 16 lanes of each half-wave
    float pr[4][8];
    float alpha[8];
#pragma unroll
    for (int r = 0; r < 8; ++r) {
      float mx = fmaxf(fmaxf(sc[0][r], sc[1][r]), fmaxf(sc[2][r], sc[3][r]));
#pragma unroll
      for (int off = 1; off < 16; off <<= 1)
        mx = fmaxf(mx, __shfl_xor(mx, off, 32));
      const float mnew = fmaxf(mrow[r], mx);
      alpha[r] = __expf(mrow[r] - mnew);
      mrow[r] = mnew;
      float rs = 0.f;
#pragma unroll
      for (int nf = 0; nf < 4; ++nf) {
        const float p = __expf(sc[nf][r] - mnew);
        pr[nf][r] = p;
        rs += p;
      }
#pragma unroll
      for (int off = 1; off < 16; off <<= 1)
        rs += __shfl_xor(rs, off, 32);
      lrow[r] = lrow[r] * alpha[r] + rs;
    }
#pragma unroll
    for (int d = 0; d < 4; ++d)
#pragma unroll
      for (int r = 0; r < 8; ++r)
        o[d][r] *= alpha[r];

    // stage P as bf16 (C-layout element -> [row][key])
#pragma unroll
    for (int nf = 0; nf < 4; ++nf)
#pragma unroll
      for (int r = 0; r < 8; ++r)
        Pw[wid][r + half * 8][nf * 16 + lan16] = f2bf(pr[nf][r]);

    __syncthreads();   // Vt + Pw visible

    // O += P @ V
#pragma unroll
    for (int d = 0; d < 4; ++d) {
#pragma unroll
      for (int kb = 0; kb < 2; ++kb) {
        Frag16 pf, vf;
        const unsigned short* pp = &Pw[wid][lan16][kb * 32 + g];
        pf.q2[0] = *(const U4*)pp;
        pf.q2[1] = *(const U4*)(pp + 16);
        const unsigned short* vp2 = &Vt[d * 16 + lan16][kb * 32 + half * 8];
        vf.q2[0] = *(const U4*)vp2;
        vf.q2[1] = *(const U4*)(vp2 + 16);
        o[d] = wmma_bf16(pf, vf, o[d]);
      }
    }
  }

  // epilogue: ctx/l + residual x -> bf16
#pragma unroll
  for (int d = 0; d < 4; ++d) {
    const int col = h * HDIM + d * 16 + lan16;
#pragma unroll
    for (int r = 0; r < 8; ++r) {
      const int srow = q0 + wid * 16 + half * 8 + r;
      const float val = o[d][r] / lrow[r];
      const size_t idx = ((size_t)b * SEQ + srow) * DIM + col;
      cb[idx] = f2bf(val + x[idx]);
    }
  }
}

// ---------------------------------------------------------------- launch
extern "C" void kernel_launch(void* const* d_in, const int* in_sizes, int n_in,
                              void* d_out, int out_size, void* d_ws, size_t ws_size,
                              hipStream_t stream) {
  const float* x  = (const float*)d_in[0];
  const float* wq = (const float*)d_in[1];
  const float* bq = (const float*)d_in[2];
  const float* wk = (const float*)d_in[3];
  const float* bk = (const float*)d_in[4];
  const float* wv = (const float*)d_in[5];
  const float* bv = (const float*)d_in[6];
  const float* wo = (const float*)d_in[7];
  const float* bo = (const float*)d_in[8];

  unsigned short* ws = (unsigned short*)d_ws;
  const size_t NX = (size_t)MROWS * DIM;   // 4 Mi halves
  const size_t NW = (size_t)DIM * DIM;     // 1 Mi halves
  unsigned short* xb  = ws;
  unsigned short* wqb = xb + NX;
  unsigned short* wkb = wqb + NW;
  unsigned short* wvb = wkb + NW;
  unsigned short* wob = wvb + NW;
  unsigned short* Qb  = wob + NW;
  unsigned short* Kb  = Qb + NX;
  unsigned short* Vb  = Kb + NX;
  unsigned short* cb  = Vb + NX;

  cvt_f32_bf16_x8<<<(int)(NX / 8 / 256), 256, 0, stream>>>(x,  xb,  (int)(NX / 8));
  cvt_f32_bf16_x8<<<(int)(NW / 8 / 256), 256, 0, stream>>>(wq, wqb, (int)(NW / 8));
  cvt_f32_bf16_x8<<<(int)(NW / 8 / 256), 256, 0, stream>>>(wk, wkb, (int)(NW / 8));
  cvt_f32_bf16_x8<<<(int)(NW / 8 / 256), 256, 0, stream>>>(wv, wvb, (int)(NW / 8));
  cvt_f32_bf16_x8<<<(int)(NW / 8 / 256), 256, 0, stream>>>(wo, wob, (int)(NW / 8));

  dim3 ggrid(DIM / BN, MROWS / BM);   // (8, 32)
  // Q carries the 1/sqrt(Hd) = 0.125 softmax scale
  gemm_bf16_wmma<0><<<ggrid, 256, 0, stream>>>(xb, wqb, bq, nullptr, Qb, 0.125f);
  gemm_bf16_wmma<0><<<ggrid, 256, 0, stream>>>(xb, wkb, bk, nullptr, Kb, 1.0f);
  gemm_bf16_wmma<0><<<ggrid, 256, 0, stream>>>(xb, wvb, bv, nullptr, Vb, 1.0f);

  dim3 agrid(SEQ / 64, BATCH * HEADS);   // (32, 32)
  attn_flash_wmma<<<agrid, 128, 0, stream>>>(Qb, Kb, Vb, x, cb);

  gemm_bf16_wmma<1><<<ggrid, 256, 0, stream>>>(cb, wob, bo, (float*)d_out,
                                               nullptr, 1.0f);
}
